// YuanAttention_33629593928083
// MI455X (gfx1250) — matrix operand order repaired and checked by reading, not compiled, not run on hardware
//
#include <hip/hip_runtime.h>
#include <math.h>

#define H 2048
#define NH 32
#define HD 64
#define S_LEN 2048

typedef _Float16 half_t;
typedef __attribute__((ext_vector_type(16))) _Float16 v16h;
typedef __attribute__((ext_vector_type(8)))  _Float16 v8h;
typedef __attribute__((ext_vector_type(8)))  float    v8f;
typedef __attribute__((ext_vector_type(4)))  unsigned int v4u;
typedef __attribute__((ext_vector_type(8)))  int      v8i;
typedef __attribute__((ext_vector_type(4)))  int      v4i;

// ---- feature detection -----------------------------------------------------
#if defined(__has_builtin)
#if __has_builtin(__builtin_amdgcn_tensor_load_to_lds) && \
    __has_builtin(__builtin_amdgcn_s_wait_tensorcnt)
#define HAVE_TDM 1
#endif
#endif
#ifndef HAVE_TDM
#define HAVE_TDM 0
#endif

#if defined(__has_include)
#if __has_include(<hip/amd_detail/amd_gfx1250_TDM.h>)
#define TDM_SIX_ARGS 1
#endif
#endif
#ifndef TDM_SIX_ARGS
#define TDM_SIX_ARGS 0
#endif

// self-report the chosen path in compile stderr
#if HAVE_TDM
#if TDM_SIX_ARGS
#pragma message("ATHENA-CFG: TDM builtin path ENABLED (six-arg form)")
#else
#pragma message("ATHENA-CFG: TDM builtin path ENABLED (five-arg form)")
#endif
#else
#pragma message("ATHENA-CFG: TDM builtin NOT available -> cooperative-load fallback")
#endif

static __device__ __forceinline__ v8f zero8() {
  v8f z;
#pragma unroll
  for (int i = 0; i < 8; ++i) z[i] = 0.0f;
  return z;
}

// ---- WMMA fragment helpers (wave32, 16x16x32 f16 -> f32) -------------------
// A fragment: 16(M) x 32(K). Lane L<16 -> row M=L, K=0..7 & 16..23;
//             lane L>=16 -> row M=L-16, K=8..15 & 24..31 (ISA 7.12.2).
static __device__ __forceinline__ v16h load_frag_a(const half_t* A,
                                                   int lda, int row0, int k0) {
  const int lane = threadIdx.x & 31;
  const int m    = lane & 15;
  const int kh   = (lane >> 4) << 3; // 0 or 8
  const half_t* p = A + (size_t)(row0 + m) * lda + k0 + kh;
  v8h lo = *(const v8h*)(p);
  v8h hi = *(const v8h*)(p + 16);
  v16h f;
#pragma unroll
  for (int i = 0; i < 8; ++i) { f[i] = lo[i]; f[i + 8] = hi[i]; }
  return f;
}

// B fragment: 32(K) x 16(N), from row-major B[N,K] (NT GEMM).
static __device__ __forceinline__ v16h load_frag_b(const half_t* B,
                                                   int ldb, int n0, int k0) {
  const int lane = threadIdx.x & 31;
  const int n    = lane & 15;
  const int ks   = (lane >> 4) << 3; // 0 or 8
  const half_t* p = B + (size_t)(n0 + n) * ldb + k0 + ks;
  v8h lo = *(const v8h*)(p);
  v8h hi = *(const v8h*)(p + 16);
  v16h f;
#pragma unroll
  for (int i = 0; i < 8; ++i) { f[i] = lo[i]; f[i + 8] = hi[i]; }
  return f;
}

static __device__ __forceinline__ v8f wmma32(v16h a, v16h b, v8f c) {
  return __builtin_amdgcn_wmma_f32_16x16x32_f16(false, a, false, b,
                                                (short)0, c, false, false);
}

// ---- Tensor Data Mover: 2D tile (tile_k x tile_rows) gmem -> LDS -----------
#if HAVE_TDM
static __device__ __forceinline__ unsigned lds_addr_of(const void* p) {
  // flat LDS aperture: low 32 bits == wave-relative LDS byte address
  return (unsigned)(uintptr_t)p;
}

static __device__ __forceinline__ void tdm_load_2d(unsigned lds_byte_addr,
                                                   const void* gptr,
                                                   unsigned tile_k,
                                                   unsigned tile_rows,
                                                   unsigned row_stride_elts) {
  const unsigned long long ga = (unsigned long long)(uintptr_t)gptr;
  v4u g0;
  g0[0] = 1u; // count=1, user flags 0, gather off
  g0[1] = (unsigned)__builtin_amdgcn_readfirstlane((int)lds_byte_addr);
  g0[2] = (unsigned)__builtin_amdgcn_readfirstlane((int)(unsigned)(ga & 0xffffffffu));
  g0[3] = (unsigned)__builtin_amdgcn_readfirstlane(
      (int)((unsigned)((ga >> 32) & 0x1ffffffu) | (2u << 30))); // type=2 (image)
  v8i g1;
  g1[0] = (int)(1u << 16);                                   // data_size=2B
  g1[1] = (int)((tile_k & 0xffffu) << 16);                   // tensor_dim0 lo
  g1[2] = (int)(((tile_k >> 16) & 0xffffu) |                 // tensor_dim0 hi
                ((tile_rows & 0xffffu) << 16));              // tensor_dim1 lo
  g1[3] = (int)(((tile_rows >> 16) & 0xffffu) |              // tensor_dim1 hi
                ((tile_k & 0xffffu) << 16));                 // tile_dim0
  g1[4] = (int)(tile_rows & 0xffffu);                        // tile_dim1 (dim2=0)
  g1[5] = (int)__builtin_amdgcn_readfirstlane((int)row_stride_elts); // stride0 lo
  g1[6] = 0;                                                 // stride0 hi, stride1 lo
  g1[7] = 0;
  v4i gz; gz[0] = 0; gz[1] = 0; gz[2] = 0; gz[3] = 0;
#if TDM_SIX_ARGS
  v8i gz8;
#pragma unroll
  for (int i = 0; i < 8; ++i) gz8[i] = 0;
  __builtin_amdgcn_tensor_load_to_lds(g0, g1, gz, gz, gz8, 0);
#else
  __builtin_amdgcn_tensor_load_to_lds(g0, g1, gz, gz, 0);
#endif
}
#endif // HAVE_TDM

// ---- NT GEMM: C[M,N] = A[M,K] * B[N,K]^T (+bias) ---------------------------
// Block: 256 thr = 8 waves; block tile 128x128; wave tile 64(M)x32(N).
// A/B K-slabs (128x32 f16) staged in LDS, double-buffered, filled by the TDM.
template <typename OutT, bool TRANS>
__global__ __launch_bounds__(256) void gemm_nt_kernel(
    const half_t* __restrict__ A, const half_t* __restrict__ B,
    OutT* __restrict__ C, const float* __restrict__ bias,
    int M, int N, int K) {
  __shared__ __align__(64) half_t Abuf[2][128 * 32];
  __shared__ __align__(64) half_t Bbuf[2][128 * 32];
  const int wave = threadIdx.x >> 5;
  const int lane = threadIdx.x & 31;
  const int wm = wave >> 2, wn = wave & 3;
  const int row_blk = blockIdx.y * 128;
  const int col_blk = blockIdx.x * 128;

  v8f acc[4][2];
#pragma unroll
  for (int i = 0; i < 4; ++i)
#pragma unroll
    for (int j = 0; j < 2; ++j) acc[i][j] = zero8();

#if HAVE_TDM
  const bool issuer = (threadIdx.x < 32); // wave-uniform
  if (issuer) {
    tdm_load_2d(lds_addr_of(&Abuf[0][0]), A + (size_t)row_blk * K, 32, 128, (unsigned)K);
    tdm_load_2d(lds_addr_of(&Bbuf[0][0]), B + (size_t)col_blk * K, 32, 128, (unsigned)K);
  }
#endif

  int ib = 0;
  for (int k0 = 0; k0 < K; k0 += 32, ib ^= 1) {
#if HAVE_TDM
    if (issuer) {
      if (k0 + 32 < K) { // prefetch next slab into the other buffer
        tdm_load_2d(lds_addr_of(&Abuf[ib ^ 1][0]),
                    A + (size_t)row_blk * K + (k0 + 32), 32, 128, (unsigned)K);
        tdm_load_2d(lds_addr_of(&Bbuf[ib ^ 1][0]),
                    B + (size_t)col_blk * K + (k0 + 32), 32, 128, (unsigned)K);
        __builtin_amdgcn_s_wait_tensorcnt(2); // current slab (2 oldest) done
      } else {
        __builtin_amdgcn_s_wait_tensorcnt(0);
      }
    }
    __syncthreads();
    const half_t* At = &Abuf[ib][0];
    const half_t* Bt = &Bbuf[ib][0];
#else
    __syncthreads(); // previous slab fully consumed
    {
      const int t2 = threadIdx.x * 2;
#pragma unroll
      for (int c = 0; c < 2; ++c) {
        const int idx = (t2 + c) * 8; // half index into 128x32 tile
        const int r = idx >> 5, kc = idx & 31;
        *(v8h*)&Abuf[0][idx] = *(const v8h*)(A + (size_t)(row_blk + r) * K + k0 + kc);
        *(v8h*)&Bbuf[0][idx] = *(const v8h*)(B + (size_t)(col_blk + r) * K + k0 + kc);
      }
    }
    __syncthreads();
    const half_t* At = &Abuf[0][0];
    const half_t* Bt = &Bbuf[0][0];
#endif
    v16h a[4], b[2];
#pragma unroll
    for (int i = 0; i < 4; ++i) a[i] = load_frag_a(At, 32, wm * 64 + i * 16, 0);
#pragma unroll
    for (int j = 0; j < 2; ++j) b[j] = load_frag_b(Bt, 32, wn * 32 + j * 16, 0);
#pragma unroll
    for (int i = 0; i < 4; ++i)
#pragma unroll
      for (int j = 0; j < 2; ++j)
        acc[i][j] = wmma32(a[i], b[j], acc[i][j]);
    __syncthreads(); // reads done before buffer is re-filled
  }

  const int n  = lane & 15;
  const int mb = (lane >> 4) * 8;
#pragma unroll
  for (int i = 0; i < 4; ++i)
#pragma unroll
    for (int j = 0; j < 2; ++j) {
      const int col = col_blk + wn * 32 + j * 16 + n;
      const float bv = bias ? bias[col] : 0.0f;
#pragma unroll
      for (int r = 0; r < 8; ++r) {
        const int row = row_blk + wm * 64 + i * 16 + mb + r;
        const float val = acc[i][j][r] + bv;
        if (TRANS) C[(size_t)col * M + row] = (OutT)val;
        else       C[(size_t)row * N + col] = (OutT)val;
      }
    }
}

// ---- Elementwise helpers ---------------------------------------------------
__global__ void cvt_f32_f16(const float* __restrict__ in, half_t* __restrict__ out,
                            int nel) {
  int i = blockIdx.x * blockDim.x + threadIdx.x;
  if (i < nel) out[i] = (half_t)in[i];
}

// conv weight [Cout, Cin, 2] -> Wcat[Cout, 2*Cin] (tap-major K blocks)
__global__ void build_wcat(const float* __restrict__ w, half_t* __restrict__ out,
                           int Cout, int Cin) {
  int i = blockIdx.x * blockDim.x + threadIdx.x;
  int tot = Cout * Cin * 2;
  if (i >= tot) return;
  int c   = i / (2 * Cin);
  int rem = i - c * 2 * Cin;
  int tap = rem / Cin;
  int ci  = rem - tap * Cin;
  out[i] = (half_t)w[((size_t)c * Cin + ci) * 2 + tap];
}

// x[S,C] -> out[S, 2C] = [x[t-1,:] (0 for t=0) , x[t,:]] in f16
template <typename T>
__global__ void build_shift_cat(const T* __restrict__ x, half_t* __restrict__ out,
                                int C) {
  int i = blockIdx.x * blockDim.x + threadIdx.x;
  int tot = S_LEN * 2 * C;
  if (i >= tot) return;
  int t = i / (2 * C);
  int k = i - t * 2 * C;
  float v;
  if (k < C) v = (t > 0) ? (float)x[(size_t)(t - 1) * C + k] : 0.0f;
  else       v = (float)x[(size_t)t * C + (k - C)];
  out[i] = (half_t)v;
}

// lf = rmsnorm(o2 + hidden) * ln_w  (f32 math, f16 out). One block per row.
__global__ __launch_bounds__(256) void lf_rmsnorm_kernel(
    const float* __restrict__ o2, const float* __restrict__ hid,
    const float* __restrict__ lnw, half_t* __restrict__ lf) {
  __shared__ float red[256];
  const int t = blockIdx.x;
  const size_t base = (size_t)t * H;
  float v[8];
  float ss = 0.0f;
#pragma unroll
  for (int i = 0; i < 8; ++i) {
    int c = threadIdx.x + i * 256;
    float x = o2[base + c] + hid[base + c];
    v[i] = x;
    ss += x * x;
  }
  red[threadIdx.x] = ss;
  __syncthreads();
  for (int o = 128; o > 0; o >>= 1) {
    if ((int)threadIdx.x < o) red[threadIdx.x] += red[threadIdx.x + o];
    __syncthreads();
  }
  const float r = rsqrtf(red[0] * (1.0f / (float)H) + 1e-6f);
#pragma unroll
  for (int i = 0; i < 8; ++i) {
    int c = threadIdx.x + i * 256;
    lf[base + c] = (half_t)(v[i] * r * lnw[c]);
  }
}

// RoPE in-place on f16 Q and K (rot_dim == HD == 64, half = 32).
__global__ __launch_bounds__(256) void rope_kernel(half_t* __restrict__ Q,
                                                   half_t* __restrict__ Km,
                                                   const int* __restrict__ pos) {
  const int t = blockIdx.x;
  const float p = (float)pos[t];
  const size_t base = (size_t)t * H;
  for (int pp = threadIdx.x; pp < H / 2; pp += blockDim.x) {
    const int hh = pp >> 5;
    const int j  = pp & 31;
    const int d0 = hh * HD + j;
    const int d1 = d0 + 32;
    const float inv = __powf(10000.0f, -(float)j * (1.0f / 32.0f));
    float sn, cs;
    __sincosf(p * inv, &sn, &cs);
    float q0 = (float)Q[base + d0], q1 = (float)Q[base + d1];
    Q[base + d0] = (half_t)(q0 * cs - q1 * sn);
    Q[base + d1] = (half_t)(q1 * cs + q0 * sn);
    float k0 = (float)Km[base + d0], k1 = (float)Km[base + d1];
    Km[base + d0] = (half_t)(k0 * cs - k1 * sn);
    Km[base + d1] = (half_t)(k1 * cs + k0 * sn);
  }
}

// ---- Flash-style causal attention ------------------------------------------
// grid = (S/128, NH), block = 256 (8 waves). Wave owns 16 query rows.
// Q,K : f16 [S, H] (head cols h*64..), Vt : f16 [H, S], O : f16 [S, H].
// K/V key-tiles (32 keys) staged in LDS by the TDM, double-buffered, shared
// by all 8 waves (vs. 8x redundant global fetch).
__global__ __launch_bounds__(256) void attn_kernel(
    const half_t* __restrict__ Q, const half_t* __restrict__ Kmat,
    const half_t* __restrict__ Vt, half_t* __restrict__ O) {
  __shared__ half_t pbuf[8][16][32]; // per-wave 16x32 P tile (C->A relayout)
#if HAVE_TDM
  __shared__ __align__(64) half_t Ktile[2][32 * 64]; // [key][d]   pitch 64
  __shared__ __align__(64) half_t Vtile[2][64 * 32]; // [d][key]   pitch 32
#endif
  const int h    = blockIdx.y;
  const int qblk = blockIdx.x * 128;
  const int wave = threadIdx.x >> 5;
  const int lane = threadIdx.x & 31;
  const int qw   = qblk + wave * 16;
  const int g = lane >> 4, n = lane & 15;

  const half_t* Qh = Q    + h * HD;
  const half_t* Kh = Kmat + h * HD;
  const half_t* Vh = Vt   + (size_t)h * HD * S_LEN;

  const v16h qa0 = load_frag_a(Qh, H, qw, 0);
  const v16h qa1 = load_frag_a(Qh, H, qw, 32);

  v8f oacc[4];
#pragma unroll
  for (int f = 0; f < 4; ++f) oacc[f] = zero8();
  float mst[8], lst[8];
#pragma unroll
  for (int r = 0; r < 8; ++r) { mst[r] = -1e30f; lst[r] = 0.0f; }

  const int kend = qblk + 128; // uniform causal bound for this block

#if HAVE_TDM
  const bool issuer = (threadIdx.x < 32);
  if (issuer) {
    tdm_load_2d(lds_addr_of(&Ktile[0][0]), Kh, 64, 32, H);
    tdm_load_2d(lds_addr_of(&Vtile[0][0]), Vh, 32, 64, S_LEN);
  }
#endif

  int ib = 0;
  for (int kk = 0; kk < kend; kk += 32, ib ^= 1) {
#if HAVE_TDM
    if (issuer) {
      if (kk + 32 < kend) {
        tdm_load_2d(lds_addr_of(&Ktile[ib ^ 1][0]), Kh + (size_t)(kk + 32) * H,
                    64, 32, H);
        tdm_load_2d(lds_addr_of(&Vtile[ib ^ 1][0]), Vh + (kk + 32),
                    32, 64, S_LEN);
        __builtin_amdgcn_s_wait_tensorcnt(2);
      } else {
        __builtin_amdgcn_s_wait_tensorcnt(0);
      }
    }
    __syncthreads();
    const half_t* Kl = &Ktile[ib][0];
    const half_t* Vl = &Vtile[ib][0];
#endif

    // S = Q * K^T for 16 queries x 32 keys (two 16x16 frags)
    v8f s0 = zero8(), s1 = zero8();
#if HAVE_TDM
    s0 = wmma32(qa0, load_frag_b(Kl, 64, 0, 0),  s0);
    s0 = wmma32(qa1, load_frag_b(Kl, 64, 0, 32), s0);
    s1 = wmma32(qa0, load_frag_b(Kl, 64, 16, 0),  s1);
    s1 = wmma32(qa1, load_frag_b(Kl, 64, 16, 32), s1);
#else
    s0 = wmma32(qa0, load_frag_b(Kh, H, kk, 0), s0);
    s0 = wmma32(qa1, load_frag_b(Kh, H, kk, 32), s0);
    s1 = wmma32(qa0, load_frag_b(Kh, H, kk + 16, 0), s1);
    s1 = wmma32(qa1, load_frag_b(Kh, H, kk + 16, 32), s1);
#endif

    float p0[8], p1[8], tmax[8];
#pragma unroll
    for (int r = 0; r < 8; ++r) {
      const int qrow = qw + g * 8 + r;
      float a = s0[r] * 0.125f; // 1/sqrt(64)
      float b = s1[r] * 0.125f;
      if (kk + n      > qrow) a = -1e30f; // causal mask
      if (kk + 16 + n > qrow) b = -1e30f;
      p0[r] = a; p1[r] = b;
      tmax[r] = fmaxf(a, b);
    }
#pragma unroll
    for (int off = 8; off >= 1; off >>= 1)
#pragma unroll
      for (int r = 0; r < 8; ++r)
        tmax[r] = fmaxf(tmax[r], __shfl_xor(tmax[r], off, 32));

    float psum[8];
#pragma unroll
    for (int r = 0; r < 8; ++r) {
      const float mnew = fmaxf(mst[r], tmax[r]);
      const float fac  = __expf(mst[r] - mnew);
      mst[r] = mnew;
      p0[r] = __expf(p0[r] - mnew);
      p1[r] = __expf(p1[r] - mnew);
      psum[r] = p0[r] + p1[r];
      lst[r] *= fac;
#pragma unroll
      for (int f = 0; f < 4; ++f) oacc[f][r] *= fac;
    }
#pragma unroll
    for (int off = 8; off >= 1; off >>= 1)
#pragma unroll
      for (int r = 0; r < 8; ++r)
        psum[r] += __shfl_xor(psum[r], off, 32);
#pragma unroll
    for (int r = 0; r < 8; ++r) lst[r] += psum[r];

    // C-layout -> A-layout through per-wave LDS tile
#pragma unroll
    for (int r = 0; r < 8; ++r) {
      pbuf[wave][g * 8 + r][n]      = (half_t)p0[r];
      pbuf[wave][g * 8 + r][16 + n] = (half_t)p1[r];
    }
    __builtin_amdgcn_wave_barrier();
    v16h pa;
    {
      const int mm = lane & 15;
      const int kh = (lane >> 4) * 8;
#pragma unroll
      for (int i = 0; i < 8; ++i) {
        pa[i]     = pbuf[wave][mm][kh + i];
        pa[i + 8] = pbuf[wave][mm][kh + 16 + i];
      }
    }
    // O += P(16x32) * V(32x64)
#pragma unroll
    for (int f = 0; f < 4; ++f)
#if HAVE_TDM
      oacc[f] = wmma32(pa, load_frag_b(Vl, 32, f * 16, 0), oacc[f]);
#else
      oacc[f] = wmma32(pa, load_frag_b(Vh, S_LEN, f * 16, kk), oacc[f]);
#endif

#if HAVE_TDM
    __syncthreads(); // all waves done with this tile before DMA refill
#endif
  }

  // epilogue: normalize and store [S, H] head-major
#pragma unroll
  for (int f = 0; f < 4; ++f)
#pragma unroll
    for (int r = 0; r < 8; ++r) {
      const int row = qw + g * 8 + r;
      const int col = h * HD + f * 16 + n;
      O[(size_t)row * H + col] = (half_t)(oacc[f][r] / lst[r]);
    }
}

// ---- Host launch -----------------------------------------------------------
extern "C" void kernel_launch(void* const* d_in, const int* in_sizes, int n_in,
                              void* d_out, int out_size, void* d_ws, size_t ws_size,
                              hipStream_t stream) {
  (void)in_sizes; (void)n_in; (void)out_size; (void)ws_size;
  const float* hid     = (const float*)d_in[0];
  const int*   pos     = (const int*)  d_in[1];
  const float* Wq      = (const float*)d_in[2];
  const float* Wk      = (const float*)d_in[3];
  const float* Wv      = (const float*)d_in[4];
  const float* Wo      = (const float*)d_in[5];
  const float* conv1_w = (const float*)d_in[6];
  const float* conv1_b = (const float*)d_in[7];
  const float* conv2_w = (const float*)d_in[8];
  const float* conv2_b = (const float*)d_in[9];
  const float* ln_w    = (const float*)d_in[10];
  float* out = (float*)d_out;

  size_t off = 0;
  auto carve = [&](size_t bytes) -> void* {
    void* p = (char*)d_ws + off;
    off += (bytes + 255) & ~(size_t)255;
    return p;
  };
  const size_t E2 = (size_t)H * H * sizeof(half_t); // 8 MB
  half_t* Wq16   = (half_t*)carve(E2);
  half_t* Wk16   = (half_t*)carve(E2);
  half_t* Wv16   = (half_t*)carve(E2);
  half_t* Wo16   = (half_t*)carve(E2);
  half_t* W1cat  = (half_t*)carve((size_t)1024 * 4096 * 2);
  half_t* W2cat  = (half_t*)carve((size_t)2048 * 2048 * 2);
  half_t* Xh16   = (half_t*)carve(E2);
  half_t* Xcat1  = (half_t*)carve((size_t)S_LEN * 4096 * 2);
  half_t* o1h    = (half_t*)carve((size_t)S_LEN * 1024 * 2);
  half_t* Xcat2  = (half_t*)carve(E2);
  float*  o2f    = (float*) carve((size_t)S_LEN * H * 4);
  half_t* lf16   = (half_t*)carve(E2);
  half_t* Q16    = (half_t*)carve(E2);
  half_t* K16    = (half_t*)carve(E2);
  half_t* Vt16   = (half_t*)carve(E2);
  half_t* attn16 = (half_t*)carve(E2);

  const int TB = 256;
  const int nE = H * H; // 4M

  // 1) precision conversion / weight packing
  cvt_f32_f16<<<nE / TB, TB, 0, stream>>>(Wq, Wq16, nE);
  cvt_f32_f16<<<nE / TB, TB, 0, stream>>>(Wk, Wk16, nE);
  cvt_f32_f16<<<nE / TB, TB, 0, stream>>>(Wv, Wv16, nE);
  cvt_f32_f16<<<nE / TB, TB, 0, stream>>>(Wo, Wo16, nE);
  cvt_f32_f16<<<nE / TB, TB, 0, stream>>>(hid, Xh16, nE);
  build_wcat<<<(1024 * 2048 * 2) / TB, TB, 0, stream>>>(conv1_w, W1cat, 1024, 2048);
  build_wcat<<<(2048 * 1024 * 2) / TB, TB, 0, stream>>>(conv2_w, W2cat, 2048, 1024);
  build_shift_cat<float><<<(S_LEN * 4096) / TB, TB, 0, stream>>>(hid, Xcat1, H);

  // 2) conv1 as GEMM: o1[2048,1024] = Xcat1[2048,4096] @ W1cat^T
  gemm_nt_kernel<half_t, false><<<dim3(1024 / 128, 2048 / 128), TB, 0, stream>>>(
      Xcat1, W1cat, o1h, conv1_b, 2048, 1024, 4096);
  build_shift_cat<half_t><<<(S_LEN * 2048) / TB, TB, 0, stream>>>(o1h, Xcat2, 1024);

  // 3) conv2 as GEMM: o2[2048,2048] (f32) = Xcat2 @ W2cat^T
  gemm_nt_kernel<float, false><<<dim3(2048 / 128, 2048 / 128), TB, 0, stream>>>(
      Xcat2, W2cat, o2f, conv2_b, 2048, 2048, 2048);

  // 4) residual + RMSNorm -> lf16
  lf_rmsnorm_kernel<<<S_LEN, TB, 0, stream>>>(o2f, hid, ln_w, lf16);

  // 5) projections: Q,K from lf ; V (transposed store) from raw hidden
  gemm_nt_kernel<half_t, false><<<dim3(16, 16), TB, 0, stream>>>(
      lf16, Wq16, Q16, nullptr, 2048, 2048, 2048);
  gemm_nt_kernel<half_t, false><<<dim3(16, 16), TB, 0, stream>>>(
      lf16, Wk16, K16, nullptr, 2048, 2048, 2048);
  gemm_nt_kernel<half_t, true><<<dim3(16, 16), TB, 0, stream>>>(
      Xh16, Wv16, Vt16, nullptr, 2048, 2048, 2048);

  // 6) RoPE (in-place, f32 math)
  rope_kernel<<<S_LEN, TB, 0, stream>>>(Q16, K16, pos);

  // 7) causal flash attention
  attn_kernel<<<dim3(S_LEN / 128, NH), TB, 0, stream>>>(Q16, K16, Vt16, attn16);

  // 8) output projection -> fp32 d_out
  gemm_nt_kernel<float, false><<<dim3(16, 16), TB, 0, stream>>>(
      attn16, Wo16, out, nullptr, 2048, 2048, 2048);
}
